// EncoderBlock_274877907645
// MI455X (gfx1250) — matrix operand order repaired and checked by reading, compiled, not run on hardware
//
#include <hip/hip_runtime.h>
#include <hip/hip_bf16.h>

// ---------------------------------------------------------------------------
// EncoderBlock for MI455X (gfx1250): bf16 WMMA GEMMs with f32 accumulate.
// B=2, S=4096, H=512, I=2048, nh=8, hd=64.
// Wave tile 32x32 (4 x v_wmma_f32_16x16x32_bf16 per k-step, 2x fragment reuse)
// with pointer-bump addressing so the hot loop is loads + WMMA only.
// ---------------------------------------------------------------------------

typedef __bf16 bf16_t;
typedef __attribute__((ext_vector_type(16))) __bf16        bf16x16;
typedef __attribute__((ext_vector_type(8)))  float         f32x8;
typedef __attribute__((ext_vector_type(8)))  unsigned int  u32x8;
typedef __attribute__((ext_vector_type(4)))  unsigned int  u32x4;

#define B_  2
#define S_  4096
#define H_  512
#define I_  2048
#define NH_ 8
#define HD_ 64
#define BS_ (B_ * S_)      // 8192 token rows
#define BH_ (B_ * NH_)     // 16 (b,h) pairs

// float -> bf16 round-to-nearest-even (no reliance on native cvt availability)
__device__ inline bf16_t f2bf(float f) {
    unsigned int u = __builtin_bit_cast(unsigned int, f);
    unsigned int r = (u + 0x7FFFu + ((u >> 16) & 1u)) >> 16;
    unsigned short h = (unsigned short)r;
    return __builtin_bit_cast(bf16_t, h);
}

__device__ inline bf16x16 mk_frag(u32x4 lo, u32x4 hi) {
    return __builtin_bit_cast(bf16x16,
        __builtin_shufflevector(lo, hi, 0, 1, 2, 3, 4, 5, 6, 7));
}

// ---------------------------------------------------------------------------
// Weight convert + transpose: W[K][N] f32 -> Wt[N][K] bf16
// ---------------------------------------------------------------------------
__global__ __launch_bounds__(256) void convt_kernel(const float* __restrict__ W,
                                                    bf16_t* __restrict__ Wt,
                                                    int K, int N) {
    long tid = (long)blockIdx.x * 256 + threadIdx.x;
    if (tid >= (long)K * N) return;
    long k = tid % K;
    long n = tid / K;
    Wt[n * (long)K + k] = f2bf(W[k * (long)N + n]);
}

// ---------------------------------------------------------------------------
// RMSNorm row kernel: one block (256 thr) per token row of H_=512.
// out bf16 = x * rsqrt(mean(x^2)+eps) * w
// ---------------------------------------------------------------------------
__global__ __launch_bounds__(256) void rmsnorm_kernel(const float* __restrict__ x,
                                                      const float* __restrict__ w,
                                                      bf16_t* __restrict__ out) {
    __shared__ float red[256];
    long row = blockIdx.x;
    const float* xr = x + row * (long)H_;
    float s = 0.f;
    for (int c = threadIdx.x; c < H_; c += 256) { float v = xr[c]; s += v * v; }
    red[threadIdx.x] = s;
    __syncthreads();
    for (int off = 128; off > 0; off >>= 1) {
        if ((int)threadIdx.x < off) red[threadIdx.x] += red[threadIdx.x + off];
        __syncthreads();
    }
    float rs = rsqrtf(red[0] / (float)H_ + 1e-5f);
    for (int c = threadIdx.x; c < H_; c += 256)
        out[row * (long)H_ + c] = f2bf(xr[c] * rs * w[c]);
}

// ---------------------------------------------------------------------------
// RoPE + per-head pack: t[BS][512] f32 -> dst[bh][s][64] bf16 (rotated)
// one thread per (row, head, d<32) pair
// ---------------------------------------------------------------------------
__global__ __launch_bounds__(256) void rope_pack_kernel(const float* __restrict__ t,
                                                        const float* __restrict__ cosT,
                                                        const float* __restrict__ sinT,
                                                        bf16_t* __restrict__ dst) {
    long tid = (long)blockIdx.x * 256 + threadIdx.x;   // BS_*256 threads total
    long row = tid >> 8;          // token row 0..8191
    int  r   = (int)(tid & 255);
    int  h   = r >> 5;            // head 0..7
    int  d   = r & 31;            // rotary dim 0..31
    long b = row >> 12;           // /4096
    long s = row & 4095;
    const float* src = t + row * (long)H_ + h * HD_;
    float x1 = src[d];
    float x2 = src[d + 32];
    float cv = cosT[s * 32 + d];
    float sv = sinT[s * 32 + d];
    bf16_t* o = dst + ((b * NH_ + h) * (long)S_ + s) * HD_;
    o[d]      = f2bf(x1 * cv - x2 * sv);
    o[d + 32] = f2bf(x2 * cv + x1 * sv);
}

// ---------------------------------------------------------------------------
// V pack + transpose: v[BS][512] f32 -> dst[bh][d][s] bf16 (Vt for PV gemm)
// ---------------------------------------------------------------------------
__global__ __launch_bounds__(256) void vpack_kernel(const float* __restrict__ t,
                                                    bf16_t* __restrict__ dst) {
    long tid = (long)blockIdx.x * 256 + threadIdx.x;   // BS_*H_ threads
    long row = tid >> 9;          // token row
    int  c   = (int)(tid & 511);
    int  h   = c >> 6;
    int  d   = c & 63;
    long b = row >> 12;
    long s = row & 4095;
    dst[((b * NH_ + h) * (long)HD_ + d) * S_ + s] = f2bf(t[tid]);
}

// ---------------------------------------------------------------------------
// Row softmax: scores f32 [rows][N] -> P bf16, one block per row (N=4096)
// ---------------------------------------------------------------------------
__global__ __launch_bounds__(256) void softmax_kernel(const float* __restrict__ S,
                                                      bf16_t* __restrict__ P,
                                                      int N) {
    __shared__ float red[256];
    long row = blockIdx.x;
    const float* sr = S + row * (long)N;
    float m = -3.0e38f;
    for (int c = threadIdx.x; c < N; c += 256) m = fmaxf(m, sr[c]);
    red[threadIdx.x] = m;
    __syncthreads();
    for (int off = 128; off > 0; off >>= 1) {
        if ((int)threadIdx.x < off)
            red[threadIdx.x] = fmaxf(red[threadIdx.x], red[threadIdx.x + off]);
        __syncthreads();
    }
    m = red[0];
    __syncthreads();
    float sum = 0.f;
    for (int c = threadIdx.x; c < N; c += 256) sum += __expf(sr[c] - m);
    red[threadIdx.x] = sum;
    __syncthreads();
    for (int off = 128; off > 0; off >>= 1) {
        if ((int)threadIdx.x < off) red[threadIdx.x] += red[threadIdx.x + off];
        __syncthreads();
    }
    float inv = 1.f / red[0];
    for (int c = threadIdx.x; c < N; c += 256)
        P[row * (long)N + c] = f2bf(__expf(sr[c] - m) * inv);
}

// ---------------------------------------------------------------------------
// Generic bf16 WMMA GEMM:  C[M][N] = A[M][K] * Bt[N][K]^T   (f32 accumulate)
// 256 threads = 8 wave32 arranged WM x WN (WM*WN == 8).
// Each wave owns a 32x32 output tile: 2 A frags x 2 B frags -> 4 WMMA/k-step,
// every loaded fragment feeds two matrix ops. Block tile = WM*32 x WN*32.
//
// Fragment addressing (per lane, half = lane/16, lm = lane%16):
//   A frag = b128 @ (rowA*2K + half*16) + {0, 32} bytes     (16x32 bf16 layout)
//   B frag = b128 @ (rowB*2K + half*16) + {0, 32} bytes     (32x16 bf16 layout)
// and every k-step advances all bases by 64 bytes -> loads + WMMA only loop.
//
// Epilogues:
//   0: outF[idx] = c*scale
//   1: outB[idx] = bf16(c*scale)
//   3: outB[idx] = bf16( silu(auxF[idx]) * c )      (fused SwiGLU)
//   4: outF[idx] = auxF[idx] + c                    (residual add)
// ---------------------------------------------------------------------------
template <int EPI, int WM, int WN>
__global__ __launch_bounds__(256) void gemm_bf16_wmma(
    const bf16_t* __restrict__ A, const bf16_t* __restrict__ Bt, int K,
    float* __restrict__ outF, bf16_t* __restrict__ outB,
    const float* __restrict__ auxF, long ldOut, float scale) {

    const int lane = threadIdx.x & 31;
    const int wave = threadIdx.x >> 5;
    const int half = lane >> 4;     // which 16-lane half
    const int lm   = lane & 15;
    const int wm   = wave % WM;
    const int wn   = wave / WM;

    const long tileM = (long)blockIdx.x * (WM * 32) + wm * 32;
    const long tileN = (long)blockIdx.y * (WN * 32) + wn * 32;

    const u32x4* pa0 = (const u32x4*)((const char*)(A  + (tileM + lm)      * (long)K) + half * 16);
    const u32x4* pa1 = (const u32x4*)((const char*)(A  + (tileM + 16 + lm) * (long)K) + half * 16);
    const u32x4* pb0 = (const u32x4*)((const char*)(Bt + (tileN + lm)      * (long)K) + half * 16);
    const u32x4* pb1 = (const u32x4*)((const char*)(Bt + (tileN + 16 + lm) * (long)K) + half * 16);

    f32x8 c00 = {}, c01 = {}, c10 = {}, c11 = {};

    for (int it = K >> 5; it > 0; --it) {
        u32x4 a0l = pa0[0], a0h = pa0[2];   // +0B, +32B
        u32x4 a1l = pa1[0], a1h = pa1[2];
        u32x4 b0l = pb0[0], b0h = pb0[2];   // +0B, +32B
        u32x4 b1l = pb1[0], b1h = pb1[2];
        pa0 += 4; pa1 += 4; pb0 += 4; pb1 += 4;   // advance 64 bytes (32 bf16)

        bf16x16 a0 = mk_frag(a0l, a0h);
        bf16x16 a1 = mk_frag(a1l, a1h);
        bf16x16 b0 = mk_frag(b0l, b0h);
        bf16x16 b1 = mk_frag(b1l, b1h);

        c00 = __builtin_amdgcn_wmma_f32_16x16x32_bf16(false, a0, false, b0,
                                                      (short)0, c00, false, false);
        c01 = __builtin_amdgcn_wmma_f32_16x16x32_bf16(false, a0, false, b1,
                                                      (short)0, c01, false, false);
        c10 = __builtin_amdgcn_wmma_f32_16x16x32_bf16(false, a1, false, b0,
                                                      (short)0, c10, false, false);
        c11 = __builtin_amdgcn_wmma_f32_16x16x32_bf16(false, a1, false, b1,
                                                      (short)0, c11, false, false);
    }

    // C fragment layout: lane -> col = lm, VGPR r -> row = r + 8*half.
#pragma unroll
    for (int mi = 0; mi < 2; ++mi) {
#pragma unroll
        for (int ni = 0; ni < 2; ++ni) {
            f32x8 c = mi ? (ni ? c11 : c10) : (ni ? c01 : c00);
            long baseRow = tileM + mi * 16 + 8 * half;
            long col     = tileN + ni * 16 + lm;
#pragma unroll
            for (int r = 0; r < 8; ++r) {
                long idx = (baseRow + r) * ldOut + col;
                float v = c[r] * scale;
                if constexpr (EPI == 0) {
                    outF[idx] = v;
                } else if constexpr (EPI == 1) {
                    outB[idx] = f2bf(v);
                } else if constexpr (EPI == 3) {
                    float g  = auxF[idx];
                    float sg = g / (1.0f + __expf(-g));   // silu(gate)
                    outB[idx] = f2bf(sg * v);
                } else if constexpr (EPI == 4) {
                    outF[idx] = auxF[idx] + v;
                }
            }
        }
    }
}

// ---------------------------------------------------------------------------
// Host orchestration
// ---------------------------------------------------------------------------
extern "C" void kernel_launch(void* const* d_in, const int* in_sizes, int n_in,
                              void* d_out, int out_size, void* d_ws, size_t ws_size,
                              hipStream_t stream) {
    (void)in_sizes; (void)n_in; (void)out_size; (void)ws_size;

    const float* x    = (const float*)d_in[0];
    const float* rcos = (const float*)d_in[1];
    const float* rsin = (const float*)d_in[2];
    const float* wq   = (const float*)d_in[3];
    const float* wk   = (const float*)d_in[4];
    const float* wv   = (const float*)d_in[5];
    const float* wo   = (const float*)d_in[6];
    const float* anw  = (const float*)d_in[7];
    const float* fnw  = (const float*)d_in[8];
    const float* wg   = (const float*)d_in[9];
    const float* wu   = (const float*)d_in[10];
    const float* wd   = (const float*)d_in[11];
    float* out = (float*)d_out;

    char*  ws  = (char*)d_ws;
    size_t off = 0;
    auto alloc = [&](size_t bytes) -> char* {
        char* p = ws + off;
        off += (bytes + 255) & ~(size_t)255;
        return p;
    };
    bf16_t* wqT  = (bf16_t*)alloc((size_t)H_ * H_ * 2);
    bf16_t* wkT  = (bf16_t*)alloc((size_t)H_ * H_ * 2);
    bf16_t* wvT  = (bf16_t*)alloc((size_t)H_ * H_ * 2);
    bf16_t* woT  = (bf16_t*)alloc((size_t)H_ * H_ * 2);
    bf16_t* wgT  = (bf16_t*)alloc((size_t)I_ * H_ * 2);
    bf16_t* wuT  = (bf16_t*)alloc((size_t)I_ * H_ * 2);
    bf16_t* wdT  = (bf16_t*)alloc((size_t)H_ * I_ * 2);
    bf16_t* xn   = (bf16_t*)alloc((size_t)BS_ * H_ * 2);    // normed activations (bf16)
    float*  tmpf = (float*) alloc((size_t)BS_ * H_ * 4);    // q/k/v pre-pack (f32)
    bf16_t* Qh   = (bf16_t*)alloc((size_t)BH_ * S_ * HD_ * 2);
    bf16_t* Kh   = (bf16_t*)alloc((size_t)BH_ * S_ * HD_ * 2);
    bf16_t* VhT  = (bf16_t*)alloc((size_t)BH_ * HD_ * S_ * 2);
    bf16_t* attO = (bf16_t*)alloc((size_t)BS_ * H_ * 2);    // attention output (pre-wo)
    float*  scor = (float*) alloc((size_t)S_ * S_ * 4);     // 64 MB, reused as gate_f32
    bf16_t* Pmat = (bf16_t*)alloc((size_t)S_ * S_ * 2);     // 32 MB, reused as ffn hidden
    float*  gatef = scor;                                   // FFN reuse (exact size match)
    bf16_t* hbuf  = Pmat;

    auto cvt = [&](const float* W, bf16_t* Wt, int K, int N) {
        long n = (long)K * N;
        convt_kernel<<<dim3((unsigned)((n + 255) / 256)), dim3(256), 0, stream>>>(W, Wt, K, N);
    };
    // Standard tiling: 8 waves as 2(M) x 4(N) -> block tile 64x128.
    auto gemm = [&](int epi, const bf16_t* A, const bf16_t* Bt, int M, int N, int K,
                    float* oF, bf16_t* oB, const float* aux, long ldOut, float scale) {
        dim3 g((unsigned)(M / 64), (unsigned)(N / 128));
        dim3 b(256);
        switch (epi) {
        case 0: gemm_bf16_wmma<0, 2, 4><<<g, b, 0, stream>>>(A, Bt, K, oF, oB, aux, ldOut, scale); break;
        case 1: gemm_bf16_wmma<1, 2, 4><<<g, b, 0, stream>>>(A, Bt, K, oF, oB, aux, ldOut, scale); break;
        case 3: gemm_bf16_wmma<3, 2, 4><<<g, b, 0, stream>>>(A, Bt, K, oF, oB, aux, ldOut, scale); break;
        case 4: gemm_bf16_wmma<4, 2, 4><<<g, b, 0, stream>>>(A, Bt, K, oF, oB, aux, ldOut, scale); break;
        }
    };
    // Narrow-N tiling for P x V (N = 64): 8 waves as 4(M) x 2(N) -> 128x64.
    auto gemm_narrow = [&](const bf16_t* A, const bf16_t* Bt, int M, int N, int K,
                           bf16_t* oB, long ldOut) {
        dim3 g((unsigned)(M / 128), (unsigned)(N / 64));
        gemm_bf16_wmma<1, 4, 2><<<g, dim3(256), 0, stream>>>(A, Bt, K, nullptr, oB,
                                                             nullptr, ldOut, 1.f);
    };

    // 1) weights -> transposed bf16
    cvt(wq, wqT, H_, H_);
    cvt(wk, wkT, H_, H_);
    cvt(wv, wvT, H_, H_);
    cvt(wo, woT, H_, H_);
    cvt(wg, wgT, H_, I_);   // [H][I] -> [I][H]
    cvt(wu, wuT, H_, I_);
    cvt(wd, wdT, I_, H_);   // [I][H] -> [H][I]

    // 2) attn RMSNorm
    rmsnorm_kernel<<<BS_, 256, 0, stream>>>(x, anw, xn);

    // 3) Q/K/V projections + RoPE + head packing (tmpf reused)
    gemm(0, xn, wqT, BS_, H_, H_, tmpf, nullptr, nullptr, H_, 1.f);
    rope_pack_kernel<<<BS_, 256, 0, stream>>>(tmpf, rcos, rsin, Qh);
    gemm(0, xn, wkT, BS_, H_, H_, tmpf, nullptr, nullptr, H_, 1.f);
    rope_pack_kernel<<<BS_, 256, 0, stream>>>(tmpf, rcos, rsin, Kh);
    gemm(0, xn, wvT, BS_, H_, H_, tmpf, nullptr, nullptr, H_, 1.f);
    vpack_kernel<<<(BS_ * H_) / 256, 256, 0, stream>>>(tmpf, VhT);

    // 4) attention per (b,h): scores + softmax + PV stay resident in 192MB L2
    const float scl = 0.125f;  // 1/sqrt(64)
    for (int bh = 0; bh < BH_; ++bh) {
        int b = bh / NH_, h = bh % NH_;
        gemm(0, Qh + (long)bh * S_ * HD_, Kh + (long)bh * S_ * HD_,
             S_, S_, HD_, scor, nullptr, nullptr, S_, scl);
        softmax_kernel<<<S_, 256, 0, stream>>>(scor, Pmat, S_);
        gemm_narrow(Pmat, VhT + (long)bh * HD_ * S_,
                    S_, HD_, S_, attO + (long)b * S_ * H_ + (long)h * HD_, H_);
    }

    // 5) output projection + residual:  out = x + attO @ wo
    gemm(4, attO, woT, BS_, H_, H_, out, nullptr, x, H_, 1.f);

    // 6) FFN RMSNorm (reuse xn)
    rmsnorm_kernel<<<BS_, 256, 0, stream>>>(out, fnw, xn);

    // 7) SwiGLU:  gate -> f32, up fused with silu(gate)*up -> bf16 hidden
    gemm(0, xn, wgT, BS_, I_, H_, gatef, nullptr, nullptr, I_, 1.f);
    gemm(3, xn, wuT, BS_, I_, H_, nullptr, hbuf, gatef, I_, 1.f);

    // 8) down projection + residual:  out += hidden @ w_down
    gemm(4, hbuf, wdT, BS_, H_, I_, out, nullptr, out, H_, 1.f);
}
